// AdvancedSimilarityComputation_77558519431269
// MI455X (gfx1250) — matrix operand order repaired and checked by reading, not compiled, not run on hardware
//
#include <hip/hip_runtime.h>
#include <hip/hip_bf16.h>

// ---------------------------------------------------------------------------
// Problem constants (match reference)
// ---------------------------------------------------------------------------
#define DIM   1024
#define NHEAD 8
#define BQ    1024
#define NC    16384
#define HDIM  128

// ---------------------------------------------------------------------------
// WMMA GEMM tile config: 128x128 block, K-step 32, 256 threads = 8 wave32
// ---------------------------------------------------------------------------
#define BM 128
#define BN 128
#define BK 32
#define LDS_STRIDE 40   // 32 + 8 bf16 pad; 80B rows keep 16B alignment

typedef __attribute__((ext_vector_type(16))) __bf16       bf16x16;
typedef __attribute__((ext_vector_type(8)))  float        floatx8;
typedef __attribute__((ext_vector_type(4)))  unsigned int uint4v;

union FragU {
    bf16x16 v;
    uint4v  q[2];
    __bf16  e[16];
};

// ---------------------------------------------------------------------------
// Tiled bf16 WMMA GEMM (NT form only):
//   C[M,N] = alpha * A[M,K] x B[N,K]^T + bias[N]
// Both A and B are row-major with contiguous K, so LDS staging is pure
// ds_store_b128 (weights are pre-transposed once on the host side below).
// Double-buffered LDS: one barrier per K-step; next tile's global_load_b128s
// are in flight across the WMMA block. Optional fp32 (Cf) and/or bf16 (Cb)
// outputs. z-dim batches via element strides (per-head attention scores).
// ---------------------------------------------------------------------------
__global__ __launch_bounds__(256)
void gemm_wmma_nt(const __bf16* __restrict__ A, const __bf16* __restrict__ Bm,
                  const float* __restrict__ bias,
                  float* __restrict__ Cf, __bf16* __restrict__ Cb,
                  int M, int N, int K, int lda, int ldb, float alpha,
                  long long strideA, long long strideB, long long strideC)
{
    __shared__ __bf16 As[2][BM * LDS_STRIDE];
    __shared__ __bf16 Bs[2][BN * LDS_STRIDE];

    A  += (long long)blockIdx.z * strideA;
    Bm += (long long)blockIdx.z * strideB;
    const long long cOff = (long long)blockIdx.z * strideC;

    const int t       = threadIdx.x;
    const int lane    = t & 31;
    const int wave    = t >> 5;        // 0..7
    const int waveRow = wave >> 1;     // 0..3 -> 32 rows each
    const int waveCol = wave & 1;      // 0..1 -> 64 cols each
    const int tileM   = blockIdx.y * BM;
    const int tileN   = blockIdx.x * BN;

    floatx8 acc[2][4];
#pragma unroll
    for (int r = 0; r < 2; ++r)
#pragma unroll
        for (int c = 0; c < 4; ++c)
#pragma unroll
            for (int v = 0; v < 8; ++v) acc[r][c][v] = 0.0f;

    // Global->LDS cooperative mapping: each thread moves 32B of A, 32B of B.
    const int arow = t >> 1, acol = (t & 1) * 16;
    const __bf16* aG = A  + (size_t)(tileM + arow) * lda + acol;
    const __bf16* bG = Bm + (size_t)(tileN + arow) * ldb + acol;
    const int sA = arow * LDS_STRIDE + acol;

    // ---- prologue: stage K-tile 0 into buffer 0
    {
        uint4v a0 = *(const uint4v*)aG;
        uint4v a1 = *(const uint4v*)(aG + 8);
        uint4v b0 = *(const uint4v*)bG;
        uint4v b1 = *(const uint4v*)(bG + 8);
        *(uint4v*)&As[0][sA]     = a0;
        *(uint4v*)&As[0][sA + 8] = a1;
        *(uint4v*)&Bs[0][sA]     = b0;
        *(uint4v*)&Bs[0][sA + 8] = b1;
    }
    __syncthreads();

    const int abase = (lane >> 4) << 3;   // A frag: K chunks at base / base+16
    const int bbase = (lane >> 4) << 3;   // B frag: K chunks at base / base+16

    int cur = 0;
    for (int k0 = 0; k0 < K; k0 += BK) {
        const bool more = (k0 + BK) < K;
        uint4v a0, a1, b0, b1;
        if (more) {   // issue next tile's loads; they fly across the WMMAs
            a0 = *(const uint4v*)(aG + k0 + BK);
            a1 = *(const uint4v*)(aG + k0 + BK + 8);
            b0 = *(const uint4v*)(bG + k0 + BK);
            b1 = *(const uint4v*)(bG + k0 + BK + 8);
            if (k0 + 2 * BK < K)  // global_prefetch_b8 two tiles ahead
                __builtin_prefetch(aG + k0 + 2 * BK, 0, 1);
        }

        // ---- fragment gathers (ISA 7.12.2 layouts) + 8 WMMAs
        const __bf16* asb = &As[cur][0];
        const __bf16* bsb = &Bs[cur][0];
        FragU afr[2], bfr[4];
#pragma unroll
        for (int r = 0; r < 2; ++r) {
            const int rowIn = waveRow * 32 + r * 16 + (lane & 15);
            const __bf16* p = asb + rowIn * LDS_STRIDE + abase;
            afr[r].q[0] = *(const uint4v*)p;
            afr[r].q[1] = *(const uint4v*)(p + 16);
        }
#pragma unroll
        for (int c = 0; c < 4; ++c) {
            const int colIn = waveCol * 64 + c * 16 + (lane & 15);
            const __bf16* p = bsb + colIn * LDS_STRIDE + bbase;
            bfr[c].q[0] = *(const uint4v*)p;
            bfr[c].q[1] = *(const uint4v*)(p + 16);
        }
#pragma unroll
        for (int r = 0; r < 2; ++r)
#pragma unroll
            for (int c = 0; c < 4; ++c)
                acc[r][c] = __builtin_amdgcn_wmma_f32_16x16x32_bf16(
                    false, afr[r].v, false, bfr[c].v,
                    (short)0, acc[r][c], false, false);

        if (more) {
            const int nxt = cur ^ 1;
            *(uint4v*)&As[nxt][sA]     = a0;
            *(uint4v*)&As[nxt][sA + 8] = a1;
            *(uint4v*)&Bs[nxt][sA]     = b0;
            *(uint4v*)&Bs[nxt][sA + 8] = b1;
            __syncthreads();          // single barrier per K-step
            cur = nxt;
        }
    }

    // ---- epilogue: C/D layout — row = (lane>=16)*8 + v, col = lane%16
    const int rb = (lane >> 4) * 8;
#pragma unroll
    for (int r = 0; r < 2; ++r) {
#pragma unroll
        for (int c = 0; c < 4; ++c) {
            const int colG = tileN + waveCol * 64 + c * 16 + (lane & 15);
            const float bv = bias ? bias[colG] : 0.0f;
#pragma unroll
            for (int v = 0; v < 8; ++v) {
                const int rowG = tileM + waveRow * 32 + r * 16 + rb + v;
                const float val = alpha * acc[r][c][v] + bv;
                const long long idx = cOff + (long long)rowG * N + colG;
                if (Cf) Cf[idx] = val;
                if (Cb) Cb[idx] = (__bf16)val;
            }
        }
    }
}

// ---------------------------------------------------------------------------
// fp32 -> bf16 convert (row-major copy, for activations)
// ---------------------------------------------------------------------------
__global__ __launch_bounds__(256)
void f32_to_bf16_kernel(const float* __restrict__ in, __bf16* __restrict__ out, int n)
{
    const int i = blockIdx.x * 256 + threadIdx.x;
    if (i < n) out[i] = (__bf16)in[i];
}

// ---------------------------------------------------------------------------
// fp32 [K,N] -> bf16 [N,K] transpose-convert (one-time, for weights).
// LDS-tiled 32x32 so both global read and write are coalesced.
// ---------------------------------------------------------------------------
__global__ __launch_bounds__(256)
void transpose_to_bf16_kernel(const float* __restrict__ in, __bf16* __restrict__ out,
                              int K, int N)
{
    __shared__ float tile[32][33];
    const int n0 = blockIdx.x * 32;
    const int k0 = blockIdx.y * 32;
    for (int i = threadIdx.y; i < 32; i += 8)
        tile[i][threadIdx.x] = in[(size_t)(k0 + i) * N + n0 + threadIdx.x];
    __syncthreads();
    for (int i = threadIdx.y; i < 32; i += 8)
        out[(size_t)(n0 + i) * K + k0 + threadIdx.x] = (__bf16)tile[threadIdx.x][i];
}

// ---------------------------------------------------------------------------
// Row LayerNorm + exact GELU, fp32 in -> bf16 out. One 256-thread WG per row.
// ---------------------------------------------------------------------------
__global__ __launch_bounds__(256)
void ln_gelu_kernel(const float* __restrict__ x, const float* __restrict__ g,
                    const float* __restrict__ be, __bf16* __restrict__ out, int D)
{
    __shared__ float red[256];
    const int row = blockIdx.x;
    const float* xr = x + (size_t)row * D;

    float s = 0.0f;
    for (int i = threadIdx.x; i < D; i += 256) s += xr[i];
    red[threadIdx.x] = s; __syncthreads();
    for (int o = 128; o > 0; o >>= 1) {
        if (threadIdx.x < o) red[threadIdx.x] += red[threadIdx.x + o];
        __syncthreads();
    }
    const float mu = red[0] / (float)D;
    __syncthreads();

    float vv = 0.0f;
    for (int i = threadIdx.x; i < D; i += 256) { float d = xr[i] - mu; vv += d * d; }
    red[threadIdx.x] = vv; __syncthreads();
    for (int o = 128; o > 0; o >>= 1) {
        if (threadIdx.x < o) red[threadIdx.x] += red[threadIdx.x + o];
        __syncthreads();
    }
    const float rstd = rsqrtf(red[0] / (float)D + 1e-5f);

    for (int i = threadIdx.x; i < D; i += 256) {
        const float y  = (xr[i] - mu) * rstd * g[i] + be[i];
        const float ge = 0.5f * y * (1.0f + erff(y * 0.70710678118654752f));
        out[(size_t)row * D + i] = (__bf16)ge;
    }
}

// ---------------------------------------------------------------------------
// Per-row sum of squares -> sq[row], inv-norm[row]
// ---------------------------------------------------------------------------
__global__ __launch_bounds__(256)
void rowsumsq_kernel(const float* __restrict__ x, float* __restrict__ sq,
                     float* __restrict__ inv, int D)
{
    __shared__ float red[256];
    const int row = blockIdx.x;
    const float* xr = x + (size_t)row * D;
    float s = 0.0f;
    for (int i = threadIdx.x; i < D; i += 256) { float v = xr[i]; s += v * v; }
    red[threadIdx.x] = s; __syncthreads();
    for (int o = 128; o > 0; o >>= 1) {
        if (threadIdx.x < o) red[threadIdx.x] += red[threadIdx.x + o];
        __syncthreads();
    }
    if (threadIdx.x == 0) {
        sq[row]  = red[0];
        inv[row] = rsqrtf(fmaxf(red[0], 1e-30f));
    }
}

// ---------------------------------------------------------------------------
// Softmax row statistics over N (per (head, batch) row): max and sum(exp(.-max))
// ---------------------------------------------------------------------------
__global__ __launch_bounds__(256)
void softmax_stats_kernel(const float* __restrict__ Th, float* __restrict__ rmax,
                          float* __restrict__ rsum, int N)
{
    __shared__ float red[256];
    const size_t base = (size_t)blockIdx.x * N;
    float m = -3.4e38f;
    for (int i = threadIdx.x; i < N; i += 256) m = fmaxf(m, Th[base + i]);
    red[threadIdx.x] = m; __syncthreads();
    for (int o = 128; o > 0; o >>= 1) {
        if (threadIdx.x < o) red[threadIdx.x] = fmaxf(red[threadIdx.x], red[threadIdx.x + o]);
        __syncthreads();
    }
    m = red[0]; __syncthreads();
    float s = 0.0f;
    for (int i = threadIdx.x; i < N; i += 256) s += expf(Th[base + i] - m);
    red[threadIdx.x] = s; __syncthreads();
    for (int o = 128; o > 0; o >>= 1) {
        if (threadIdx.x < o) red[threadIdx.x] += red[threadIdx.x + o];
        __syncthreads();
    }
    if (threadIdx.x == 0) { rmax[blockIdx.x] = m; rsum[blockIdx.x] = red[0]; }
}

// ---------------------------------------------------------------------------
// Final fusion: cosine / euclidean / learned similarity -> MLP(3->64->1) -> sigmoid
// ---------------------------------------------------------------------------
__global__ __launch_bounds__(256)
void fuse_kernel(const float* __restrict__ S, const float* __restrict__ Th,
                 const float* __restrict__ rmax, const float* __restrict__ rsum,
                 const float* __restrict__ qsq, const float* __restrict__ qinv,
                 const float* __restrict__ ksq, const float* __restrict__ kinv,
                 const float* __restrict__ temp,
                 const float* __restrict__ fw1, const float* __restrict__ fb1,
                 const float* __restrict__ fw2, const float* __restrict__ fb2,
                 float* __restrict__ out, int Bn, int Nn, int Hn)
{
    __shared__ float w1s[192], b1s[64], w2s[64];
    const int t = threadIdx.x;
    if (t < 192) w1s[t] = fw1[t];
    if (t < 64) { b1s[t] = fb1[t]; w2s[t] = fw2[t]; }
    __syncthreads();

    const int b = blockIdx.y;
    const int n = blockIdx.x * 256 + t;

    const float expT = expf(temp[0]);
    const float s    = S[(size_t)b * Nn + n];
    const float cosv = s * qinv[b] * kinv[n] * expT;

    float d2 = fmaxf(qsq[b] + ksq[n] - 2.0f * s, 0.0f);
    const float euc = 1.0f / (1.0f + sqrtf(d2));

    float l = 0.0f;
    for (int h = 0; h < Hn; ++h) {
        const int r = h * Bn + b;
        l += expf(Th[(size_t)r * Nn + n] - rmax[r]) / rsum[r];
    }
    l *= (1.0f / (float)Hn);

    float accv = fb2[0];
#pragma unroll
    for (int j = 0; j < 64; ++j) {
        const float hj = fmaf(cosv, w1s[j],
                         fmaf(euc,  w1s[64 + j],
                         fmaf(l,    w1s[128 + j], b1s[j])));
        accv = fmaf(fmaxf(hj, 0.0f), w2s[j], accv);
    }
    out[(size_t)b * Nn + n] = 1.0f / (1.0f + expf(-accv));
}

// ---------------------------------------------------------------------------
// Host orchestration
// ---------------------------------------------------------------------------
extern "C" void kernel_launch(void* const* d_in, const int* in_sizes, int n_in,
                              void* d_out, int out_size, void* d_ws, size_t ws_size,
                              hipStream_t stream)
{
    (void)in_sizes; (void)n_in; (void)out_size; (void)ws_size;

    const float* Xq   = (const float*)d_in[0];
    const float* Xk   = (const float*)d_in[1];
    const float* temp = (const float*)d_in[2];
    const float* q_w1 = (const float*)d_in[3];
    const float* q_b1 = (const float*)d_in[4];
    const float* q_g  = (const float*)d_in[5];
    const float* q_be = (const float*)d_in[6];
    const float* q_w2 = (const float*)d_in[7];
    const float* q_b2 = (const float*)d_in[8];
    const float* k_w1 = (const float*)d_in[9];
    const float* k_b1 = (const float*)d_in[10];
    const float* k_g  = (const float*)d_in[11];
    const float* k_be = (const float*)d_in[12];
    const float* k_w2 = (const float*)d_in[13];
    const float* k_b2 = (const float*)d_in[14];
    const float* wq   = (const float*)d_in[15];
    const float* bq   = (const float*)d_in[16];
    const float* wk   = (const float*)d_in[17];
    const float* bk   = (const float*)d_in[18];
    const float* fw1  = (const float*)d_in[19];
    const float* fb1  = (const float*)d_in[20];
    const float* fw2  = (const float*)d_in[21];
    const float* fb2  = (const float*)d_in[22];
    float* out        = (float*)d_out;

    // workspace carve-out (256B aligned)
    char*  ws  = (char*)d_ws;
    size_t off = 0;
    auto alloc = [&](size_t bytes) -> void* {
        void* p = ws + off;
        off = (off + bytes + 255) & ~(size_t)255;
        return p;
    };
    const size_t DD = (size_t)DIM * DIM;

    __bf16* qw1t = (__bf16*)alloc(DD * 2);     // transposed bf16 weights [N,K]
    __bf16* qw2t = (__bf16*)alloc(DD * 2);
    __bf16* kw1t = (__bf16*)alloc(DD * 2);
    __bf16* kw2t = (__bf16*)alloc(DD * 2);
    __bf16* wqt  = (__bf16*)alloc(DD * 2);
    __bf16* wkt  = (__bf16*)alloc(DD * 2);
    __bf16* xqb  = (__bf16*)alloc((size_t)BQ * DIM * 2);
    __bf16* xkb  = (__bf16*)alloc((size_t)NC * DIM * 2);
    float*  t1   = (float*) alloc((size_t)NC * DIM * 4);   // reused: tq1 then tk1
    __bf16* hqb  = (__bf16*)alloc((size_t)BQ * DIM * 2);
    __bf16* hkb  = (__bf16*)alloc((size_t)NC * DIM * 2);
    float*  qp   = (float*) alloc((size_t)BQ * DIM * 4);
    __bf16* qpb  = (__bf16*)alloc((size_t)BQ * DIM * 2);
    float*  kp   = (float*) alloc((size_t)NC * DIM * 4);
    __bf16* kpb  = (__bf16*)alloc((size_t)NC * DIM * 2);
    __bf16* qhb  = (__bf16*)alloc((size_t)BQ * DIM * 2);
    __bf16* khb  = (__bf16*)alloc((size_t)NC * DIM * 2);
    float*  Smat = (float*) alloc((size_t)BQ * NC * 4);
    float*  Th   = (float*) alloc((size_t)NHEAD * BQ * NC * 4);
    float*  qsq  = (float*) alloc((size_t)BQ * 4);
    float*  qinv = (float*) alloc((size_t)BQ * 4);
    float*  ksq  = (float*) alloc((size_t)NC * 4);
    float*  kinv = (float*) alloc((size_t)NC * 4);
    float*  rmax = (float*) alloc((size_t)NHEAD * BQ * 4);
    float*  rsum = (float*) alloc((size_t)NHEAD * BQ * 4);

    const dim3 blk(256);
    const dim3 tblk(32, 8);
    const dim3 tgrd(DIM / 32, DIM / 32);

    // ---- one-time weight transpose+convert to bf16 [N,K]
    transpose_to_bf16_kernel<<<tgrd, tblk, 0, stream>>>(q_w1, qw1t, DIM, DIM);
    transpose_to_bf16_kernel<<<tgrd, tblk, 0, stream>>>(q_w2, qw2t, DIM, DIM);
    transpose_to_bf16_kernel<<<tgrd, tblk, 0, stream>>>(k_w1, kw1t, DIM, DIM);
    transpose_to_bf16_kernel<<<tgrd, tblk, 0, stream>>>(k_w2, kw2t, DIM, DIM);
    transpose_to_bf16_kernel<<<tgrd, tblk, 0, stream>>>(wq,   wqt,  DIM, DIM);
    transpose_to_bf16_kernel<<<tgrd, tblk, 0, stream>>>(wk,   wkt,  DIM, DIM);

    // ---- activation converts
    f32_to_bf16_kernel<<<(BQ * DIM) / 256, blk, 0, stream>>>(Xq, xqb, BQ * DIM);
    f32_to_bf16_kernel<<<(NC * DIM) / 256, blk, 0, stream>>>(Xk, xkb, NC * DIM);

    // ---- query projection: GEMM1 -> LN+GELU -> GEMM2 (keep fp32 + bf16)
    gemm_wmma_nt<<<dim3(DIM / BN, BQ / BM, 1), blk, 0, stream>>>(
        xqb, qw1t, q_b1, t1, nullptr, BQ, DIM, DIM, DIM, DIM, 1.0f, 0, 0, 0);
    ln_gelu_kernel<<<BQ, blk, 0, stream>>>(t1, q_g, q_be, hqb, DIM);
    gemm_wmma_nt<<<dim3(DIM / BN, BQ / BM, 1), blk, 0, stream>>>(
        hqb, qw2t, q_b2, qp, qpb, BQ, DIM, DIM, DIM, DIM, 1.0f, 0, 0, 0);

    // ---- key projection
    gemm_wmma_nt<<<dim3(DIM / BN, NC / BM, 1), blk, 0, stream>>>(
        xkb, kw1t, k_b1, t1, nullptr, NC, DIM, DIM, DIM, DIM, 1.0f, 0, 0, 0);
    ln_gelu_kernel<<<NC, blk, 0, stream>>>(t1, k_g, k_be, hkb, DIM);
    gemm_wmma_nt<<<dim3(DIM / BN, NC / BM, 1), blk, 0, stream>>>(
        hkb, kw2t, k_b2, kp, kpb, NC, DIM, DIM, DIM, DIM, 1.0f, 0, 0, 0);

    // ---- MHA in-projections (bf16 only)
    gemm_wmma_nt<<<dim3(DIM / BN, BQ / BM, 1), blk, 0, stream>>>(
        qpb, wqt, bq, nullptr, qhb, BQ, DIM, DIM, DIM, DIM, 1.0f, 0, 0, 0);
    gemm_wmma_nt<<<dim3(DIM / BN, NC / BM, 1), blk, 0, stream>>>(
        kpb, wkt, bk, nullptr, khb, NC, DIM, DIM, DIM, DIM, 1.0f, 0, 0, 0);

    // ---- row norms / squared norms
    rowsumsq_kernel<<<BQ, blk, 0, stream>>>(qp, qsq, qinv, DIM);
    rowsumsq_kernel<<<NC, blk, 0, stream>>>(kp, ksq, kinv, DIM);

    // ---- S = qp @ kp^T  (serves cosine + euclidean); kpb is [NC, DIM]
    gemm_wmma_nt<<<dim3(NC / BN, BQ / BM, 1), blk, 0, stream>>>(
        qpb, kpb, nullptr, Smat, nullptr, BQ, NC, DIM, DIM, DIM, 1.0f, 0, 0, 0);

    // ---- per-head scores Th[h,b,n] = (qh_h @ kh_h^T) / sqrt(HD), batched over z
    gemm_wmma_nt<<<dim3(NC / BN, BQ / BM, NHEAD), blk, 0, stream>>>(
        qhb, khb, nullptr, Th, nullptr, BQ, NC, HDIM, DIM, DIM,
        0.08838834764831845f, (long long)HDIM, (long long)HDIM,
        (long long)BQ * NC);

    // ---- softmax statistics per (h,b) row
    softmax_stats_kernel<<<NHEAD * BQ, blk, 0, stream>>>(Th, rmax, rsum, NC);

    // ---- final fusion MLP
    fuse_kernel<<<dim3(NC / 256, BQ, 1), blk, 0, stream>>>(
        Smat, Th, rmax, rsum, qsq, qinv, ksq, kinv, temp,
        fw1, fb1, fw2, fb2, out, BQ, NC, NHEAD);
}
